// OGBEmbedCINpp_53085795779156
// MI455X (gfx1250) — matrix operand order repaired and checked by reading, not compiled, not run on hardware
//
#include <hip/hip_runtime.h>
#include <hip/hip_bf16.h>

// ---------------------------------------------------------------------------
// CIN++ forward for MI455X (gfx1250, wave32, WMMA).
// All matmuls run through v_wmma_f32_16x16x32_bf16 (bf16 A/B, f32 accum).
// Weights are repacked per call into fragment-major bf16 (one contiguous
// 32-byte v16bf per lane per 16x16x32 fragment). Each block now processes
// 32 rows: every wave keeps two A fragments and four hoisted B fragments in
// registers and issues 8 back-to-back WMMAs per k-step (one wait per burst),
// prefetching the next k-step's B fragments (global_prefetch_b8).
// Segment sums use global f32 atomics into aggregates pre-seeded with
// (1+eps)*x; training-mode BN stats via column-reduction kernels, folded
// into the next GEMM's A-conversion as y*scale+shift (+ReLU).
// ---------------------------------------------------------------------------

typedef __attribute__((ext_vector_type(16))) __bf16 v16bf;
typedef __attribute__((ext_vector_type(8)))  __bf16 v8bf;
typedef __attribute__((ext_vector_type(8)))  float  v8f;

#define NCELL 40000
#define NBND  60000
#define NEU   200000
#define NED   200000
#define NEB   120000

static __device__ __forceinline__ __bf16 f2bf(float f) {
    union { float f; unsigned u; } a; a.f = f;
    unsigned r = a.u + 0x7FFFu + ((a.u >> 16) & 1u);   // round-to-nearest-even
    unsigned short h = (unsigned short)(r >> 16);
    union { unsigned short s; __bf16 b; } c; c.s = h;
    return c.b;
}

static __device__ __forceinline__ v16bf wmma_bf16(v16bf a, v16bf b, v8f& acc) {
    acc = __builtin_amdgcn_wmma_f32_16x16x32_bf16(
        false, a, false, b, (short)0, acc, false, false);
    return a;
}

// ---- Repack a KxNcols f32 row-major weight into WMMA fragment-major bf16 ---
// Fragment (kt, nt) covers K rows [kt*32, kt*32+32) x cols [nt*16, nt*16+16).
// Lane l (0..31): col = nt*16 + (l&15); elem e<8: k = kt*32 + (l>>4)*8 + e, e>=8: k = kt*32 + 16 + (l>>4)*8 + (e-8).
__global__ void repack_w(const float* __restrict__ W, __bf16* __restrict__ out,
                         int K, int Ncols) {
    int idx = blockIdx.x * blockDim.x + threadIdx.x;
    if (idx >= K * Ncols) return;
    int frag   = idx >> 9;        // 512 elems (32 lanes x 16) per fragment
    int within = idx & 511;
    int lane   = within >> 4;
    int e      = within & 15;
    int ntiles = Ncols >> 4;
    int nt  = frag % ntiles;
    int kt  = frag / ntiles;
    int col = nt * 16 + (lane & 15);
    int k   = kt * 32 + ((e >> 3) << 4) + ((lane >> 4) << 3) + (e & 7);
    out[idx] = f2bf(W[(size_t)k * Ncols + col]);
}

// ---- Initialize the three aggregates to (1+eps_i) * x -----------------------
__global__ void init_bufs(const float* __restrict__ x,
                          const float* e1, const float* e2, const float* e3,
                          float* __restrict__ bu, float* __restrict__ bd,
                          float* __restrict__ bb) {
    size_t i = (size_t)blockIdx.x * 256 + threadIdx.x;
    float v = x[i];
    bu[i] = (1.0f + e1[0]) * v;
    bd[i] = (1.0f + e2[0]) * v;
    bb[i] = (1.0f + e3[0]) * v;
}

__global__ void init_ident(float* sid, float* hid) {
    sid[threadIdx.x] = 1.0f; hid[threadIdx.x] = 0.0f;
}

__global__ void zero2(float* a, float* b) {
    a[threadIdx.x] = 0.0f; b[threadIdx.x] = 0.0f;
}

// ---- Edge message GEMM + ReLU + atomic scatter ------------------------------
// Block = 128 threads (4 waves) handles 32 edges. A = 32 x 512 bf16 in LDS
// (cols 0..255 = x[src], 256..511 = attr[e]). B = packed W (512x256).
// Each wave owns 4 column tiles x 2 row tiles = 8 accumulators.
__global__ __launch_bounds__(128)
void edge_msg(const float* __restrict__ x, const float* __restrict__ attr,
              const int* __restrict__ src, const int* __restrict__ tgt,
              const __bf16* __restrict__ Bp, const float* __restrict__ bias,
              float* __restrict__ agg) {
    __shared__ __bf16 lA[32 * 512];
    __shared__ int ssrc[32], stgt[32];
    const int tid = threadIdx.x;
    const int e0  = blockIdx.x * 32;
    if (tid < 32) { ssrc[tid] = src[e0 + tid]; stgt[tid] = tgt[e0 + tid]; }
    __syncthreads();

    for (int idx = tid * 4; idx < 32 * 512; idx += 128 * 4) {
        int row = idx >> 9;
        int col = idx & 511;
        const float* gp = (col < 256)
            ? (x    + (size_t)ssrc[row] * 256 + col)
            : (attr + (size_t)(e0 + row) * 256 + (col - 256));
        float4 v = *(const float4*)gp;
        __bf16* lp = &lA[idx];
        lp[0] = f2bf(v.x); lp[1] = f2bf(v.y); lp[2] = f2bf(v.z); lp[3] = f2bf(v.w);
    }
    __syncthreads();

    const int lane = tid & 31;
    const int wave = tid >> 5;
    const int rowA = lane & 15;
    const int koff = (lane >> 4) << 3;

    v8f acc[2][4] = {};
    for (int kt = 0; kt < 16; ++kt) {
        v16bf a[2];
#pragma unroll
        for (int r = 0; r < 2; ++r) {
            const __bf16* ap = &lA[(rowA + r * 16) * 512 + kt * 32 + koff];
            v8bf lo = *(const v8bf*)ap;
            v8bf hi = *(const v8bf*)(ap + 16);
#pragma unroll
            for (int i = 0; i < 8; ++i) { a[r][i] = lo[i]; a[r][8 + i] = hi[i]; }
        }
        v16bf b[4];
#pragma unroll
        for (int j = 0; j < 4; ++j) {
            int nt = wave + j * 4;
            b[j] = *(const v16bf*)(Bp + ((size_t)(kt * 16 + nt) * 32 + lane) * 16);
        }
        if (kt + 1 < 16)
            __builtin_prefetch(Bp + ((size_t)((kt + 1) * 16 + wave) * 32 + lane) * 16, 0, 3);
#pragma unroll
        for (int j = 0; j < 4; ++j) {
            wmma_bf16(a[0], b[j], acc[0][j]);
            wmma_bf16(a[1], b[j], acc[1][j]);
        }
    }
#pragma unroll
    for (int j = 0; j < 4; ++j) {
        int col = (wave + j * 4) * 16 + (lane & 15);
        float bs = bias[col];
#pragma unroll
        for (int r = 0; r < 2; ++r) {
#pragma unroll
            for (int v = 0; v < 8; ++v) {
                int rl = r * 16 + ((lane < 16) ? v : (v + 8));
                float val = acc[r][j][v] + bs;
                val = val > 0.0f ? val : 0.0f;
                atomicAdd(&agg[(size_t)stgt[rl] * 256 + col], val);
            }
        }
    }
}

// ---- Boundary gather + atomic scatter (identity message) --------------------
__global__ __launch_bounds__(64)
void bnd_scatter(const float* __restrict__ battr, const int* __restrict__ src,
                 const int* __restrict__ tgt, float* __restrict__ agg) {
    int e = blockIdx.x;
    int s = src[e], t = tgt[e];
    int c = threadIdx.x * 4;
    float4 v = *(const float4*)(battr + (size_t)s * 256 + c);
    float* p = agg + (size_t)t * 256 + c;
    atomicAdd(p + 0, v.x); atomicAdd(p + 1, v.y);
    atomicAdd(p + 2, v.z); atomicAdd(p + 3, v.w);
}

// ---- Generic node GEMM: Y = normReLU(A) @ W + bias --------------------------
// A is 1..3 segments of 256 cols, each with its own BN scale/shift; ReLU on
// conversion when relu_in != 0. 32 rows per block; in-place safe per block.
__global__ __launch_bounds__(128)
void gemm_mlp(const float* __restrict__ A0, const float* __restrict__ A1,
              const float* __restrict__ A2,
              const float* __restrict__ sc0, const float* __restrict__ sh0,
              const float* __restrict__ sc1, const float* __restrict__ sh1,
              const float* __restrict__ sc2, const float* __restrict__ sh2,
              int nseg, int relu_in,
              const __bf16* __restrict__ Bp, const float* __restrict__ bias,
              float* __restrict__ Y) {
    __shared__ __bf16 lA[32 * 768];
    const int tid = threadIdx.x;
    const int K   = nseg << 8;
    const int m0  = blockIdx.x * 32;
    const float* As[3]  = {A0, A1, A2};
    const float* scs[3] = {sc0, sc1, sc2};
    const float* shs[3] = {sh0, sh1, sh2};

    for (int idx = tid * 4; idx < 32 * K; idx += 128 * 4) {
        int row = idx / K;
        int col = idx - row * K;
        int seg = col >> 8;
        int c   = col & 255;
        float4 v = *(const float4*)(As[seg] + (size_t)(m0 + row) * 256 + c);
        const float* sc = scs[seg];
        const float* sh = shs[seg];
        float f[4] = {v.x, v.y, v.z, v.w};
#pragma unroll
        for (int j = 0; j < 4; ++j) {
            float t = f[j] * sc[c + j] + sh[c + j];
            if (relu_in) t = t > 0.0f ? t : 0.0f;
            lA[idx + j] = f2bf(t);
        }
    }
    __syncthreads();

    const int lane = tid & 31;
    const int wave = tid >> 5;
    const int rowA = lane & 15;
    const int koff = (lane >> 4) << 3;
    const int nkt  = K >> 5;

    v8f acc[2][4] = {};
    for (int kt = 0; kt < nkt; ++kt) {
        v16bf a[2];
#pragma unroll
        for (int r = 0; r < 2; ++r) {
            const __bf16* ap = &lA[(rowA + r * 16) * K + kt * 32 + koff];
            v8bf lo = *(const v8bf*)ap;
            v8bf hi = *(const v8bf*)(ap + 16);
#pragma unroll
            for (int i = 0; i < 8; ++i) { a[r][i] = lo[i]; a[r][8 + i] = hi[i]; }
        }
        v16bf b[4];
#pragma unroll
        for (int j = 0; j < 4; ++j) {
            int nt = wave + j * 4;
            b[j] = *(const v16bf*)(Bp + ((size_t)(kt * 16 + nt) * 32 + lane) * 16);
        }
        if (kt + 1 < nkt)
            __builtin_prefetch(Bp + ((size_t)((kt + 1) * 16 + wave) * 32 + lane) * 16, 0, 3);
#pragma unroll
        for (int j = 0; j < 4; ++j) {
            wmma_bf16(a[0], b[j], acc[0][j]);
            wmma_bf16(a[1], b[j], acc[1][j]);
        }
    }
#pragma unroll
    for (int j = 0; j < 4; ++j) {
        int col = (wave + j * 4) * 16 + (lane & 15);
        float bs = bias[col];
#pragma unroll
        for (int r = 0; r < 2; ++r) {
#pragma unroll
            for (int v = 0; v < 8; ++v) {
                int rl = r * 16 + ((lane < 16) ? v : (v + 8));
                Y[(size_t)(m0 + rl) * 256 + col] = acc[r][j][v] + bs;
            }
        }
    }
}

// ---- Column partial sums for BN stats ---------------------------------------
__global__ void col_partial(const float* __restrict__ Y, int nrows, int rpb,
                            float* __restrict__ cs, float* __restrict__ cq) {
    int c  = threadIdx.x;
    int r0 = blockIdx.x * rpb;
    int r1 = r0 + rpb; if (r1 > nrows) r1 = nrows;
    float s = 0.0f, q = 0.0f;
    for (int r = r0; r < r1; ++r) {
        float v = Y[(size_t)r * 256 + c];
        s += v; q += v * v;
    }
    atomicAdd(&cs[c], s);
    atomicAdd(&cq[c], q);
}

__global__ void fin_stats(const float* cs, const float* cq,
                          const float* __restrict__ g, const float* __restrict__ b,
                          float invn, float* sc, float* sh) {
    int t = threadIdx.x;
    float mean = cs[t] * invn;
    float var  = cq[t] * invn - mean * mean;
    float rstd = rsqrtf(var + 1e-5f);
    float s = g[t] * rstd;
    sc[t] = s;
    sh[t] = b[t] - s * mean;
}

__global__ void final_out(const float* __restrict__ Y, const float* sc,
                          const float* sh, float* __restrict__ out) {
    size_t i = (size_t)blockIdx.x * 256 + threadIdx.x;
    int c = threadIdx.x;
    float v = Y[i] * sc[c] + sh[c];
    out[i] = v > 0.0f ? v : 0.0f;
}

// ---------------------------------------------------------------------------
extern "C" void kernel_launch(void* const* d_in, const int* in_sizes, int n_in,
                              void* d_out, int out_size, void* d_ws, size_t ws_size,
                              hipStream_t stream) {
    (void)in_sizes; (void)n_in; (void)out_size; (void)ws_size;

    const float* x         = (const float*)d_in[0];
    const float* up_attr   = (const float*)d_in[1];
    const float* down_attr = (const float*)d_in[2];
    const float* bnd_attr  = (const float*)d_in[3];
    const float* Wmu = (const float*)d_in[4];  const float* bmu = (const float*)d_in[5];
    const float* Wmd = (const float*)d_in[6];  const float* bmd = (const float*)d_in[7];
    const float *W1[3], *b1[3], *g1[3], *c1[3], *W2[3], *b2[3], *g2[3], *c2[3];
    for (int t = 0; t < 3; ++t) {
        int base = 8 + t * 8;
        W1[t] = (const float*)d_in[base + 0]; b1[t] = (const float*)d_in[base + 1];
        g1[t] = (const float*)d_in[base + 2]; c1[t] = (const float*)d_in[base + 3];
        W2[t] = (const float*)d_in[base + 4]; b2[t] = (const float*)d_in[base + 5];
        g2[t] = (const float*)d_in[base + 6]; c2[t] = (const float*)d_in[base + 7];
    }
    const float* Wc = (const float*)d_in[32]; const float* bc = (const float*)d_in[33];
    const float* gc = (const float*)d_in[34]; const float* cc = (const float*)d_in[35];
    const float* e1 = (const float*)d_in[36];
    const float* e2 = (const float*)d_in[37];
    const float* e3 = (const float*)d_in[38];
    const int* up_idx = (const int*)d_in[39];
    const int* dn_idx = (const int*)d_in[40];
    const int* bd_idx = (const int*)d_in[41];

    // ---- workspace layout ----
    const size_t nelt = (size_t)NCELL * 256;
    float* buf[3];
    buf[0] = (float*)d_ws;
    buf[1] = buf[0] + nelt;
    buf[2] = buf[1] + nelt;
    __bf16* wp_mu = (__bf16*)(buf[2] + nelt);
    __bf16* wp_md = wp_mu + 512 * 256;
    __bf16* wp1[3], *wp2[3];
    wp1[0] = wp_md + 512 * 256;
    wp1[1] = wp1[0] + 256 * 256; wp1[2] = wp1[1] + 256 * 256;
    wp2[0] = wp1[2] + 256 * 256;
    wp2[1] = wp2[0] + 256 * 256; wp2[2] = wp2[1] + 256 * 256;
    __bf16* wp_c = wp2[2] + 256 * 256;
    float* st  = (float*)(wp_c + 768 * 256);
    float* cs  = st;        float* cq  = st + 256;
    float* sid = st + 512;  float* hid = st + 768;
    float* s1  = st + 1024; float* h1  = st + 1280;
    float* s2[3], *h2[3];
    s2[0] = st + 1536; h2[0] = st + 1792;
    s2[1] = st + 2048; h2[1] = st + 2304;
    s2[2] = st + 2560; h2[2] = st + 2816;
    float* sC = st + 3072;  float* hC = st + 3328;

    // ---- repack weights to WMMA fragment layout (bf16) ----
    repack_w<<<(512 * 256 + 255) / 256, 256, 0, stream>>>(Wmu, wp_mu, 512, 256);
    repack_w<<<(512 * 256 + 255) / 256, 256, 0, stream>>>(Wmd, wp_md, 512, 256);
    for (int t = 0; t < 3; ++t) {
        repack_w<<<(256 * 256 + 255) / 256, 256, 0, stream>>>(W1[t], wp1[t], 256, 256);
        repack_w<<<(256 * 256 + 255) / 256, 256, 0, stream>>>(W2[t], wp2[t], 256, 256);
    }
    repack_w<<<(768 * 256 + 255) / 256, 256, 0, stream>>>(Wc, wp_c, 768, 256);

    init_ident<<<1, 256, 0, stream>>>(sid, hid);
    init_bufs<<<NCELL, 256, 0, stream>>>(x, e1, e2, e3, buf[0], buf[1], buf[2]);

    // ---- message passing ----
    edge_msg<<<NEU / 32, 128, 0, stream>>>(x, up_attr, up_idx, up_idx + NEU,
                                           wp_mu, bmu, buf[0]);
    edge_msg<<<NED / 32, 128, 0, stream>>>(x, down_attr, dn_idx, dn_idx + NED,
                                           wp_md, bmd, buf[1]);
    bnd_scatter<<<NEB, 64, 0, stream>>>(bnd_attr, bd_idx, bd_idx + NEB, buf[2]);

    const float invN = 1.0f / (float)NCELL;
    const int mblocks = NCELL / 32;
    const int sblocks = (NCELL + 255) / 256;

    // ---- per-branch MLPs (Linear -> BN -> ReLU -> Linear -> BN -> ReLU) ----
    for (int t = 0; t < 3; ++t) {
        gemm_mlp<<<mblocks, 128, 0, stream>>>(buf[t], nullptr, nullptr,
            sid, hid, sid, hid, sid, hid, 1, 0, wp1[t], b1[t], buf[t]);
        zero2<<<1, 256, 0, stream>>>(cs, cq);
        col_partial<<<sblocks, 256, 0, stream>>>(buf[t], NCELL, 256, cs, cq);
        fin_stats<<<1, 256, 0, stream>>>(cs, cq, g1[t], c1[t], invN, s1, h1);

        gemm_mlp<<<mblocks, 128, 0, stream>>>(buf[t], nullptr, nullptr,
            s1, h1, sid, hid, sid, hid, 1, 1, wp2[t], b2[t], buf[t]);
        zero2<<<1, 256, 0, stream>>>(cs, cq);
        col_partial<<<sblocks, 256, 0, stream>>>(buf[t], NCELL, 256, cs, cq);
        fin_stats<<<1, 256, 0, stream>>>(cs, cq, g2[t], c2[t], invN, s2[t], h2[t]);
    }

    // ---- concat (as 3 K-segments) -> Linear -> BN -> ReLU ----
    gemm_mlp<<<mblocks, 128, 0, stream>>>(buf[0], buf[1], buf[2],
        s2[0], h2[0], s2[1], h2[1], s2[2], h2[2], 3, 1, wp_c, bc, buf[0]);
    zero2<<<1, 256, 0, stream>>>(cs, cq);
    col_partial<<<sblocks, 256, 0, stream>>>(buf[0], NCELL, 256, cs, cq);
    fin_stats<<<1, 256, 0, stream>>>(cs, cq, gc, cc, invN, sC, hC);

    final_out<<<NCELL, 256, 0, stream>>>(buf[0], sC, hC, (float*)d_out);
}
